// Wavelet2D_45183055954445
// MI455X (gfx1250) — matrix operand order repaired and checked by reading, not compiled, not run on hardware
//
#include <hip/hip_runtime.h>
#include <hip/hip_bf16.h>

// ---------------- problem constants ----------------
#define B_   4
#define C_   32
#define H_   192
#define W_   192
#define HW_  (H_*W_)
#define BC_  (B_*C_)     // 128
#define K_   6
#define WO_  190         // even-length FIR conv output size (192-2)
#define HN_  96          // lifting half resolution
#define CIN_ 128         // 4*C channels into projection
#define COUT_ 32

// Workspace layout (floats). Peak ~190.3 MB — nearly L2 resident on MI455X.
#define OFF_GATE 0
#define SZ_GATE  (B_*K_*HW_)               // 884736
#define OFF_A    (OFF_GATE + SZ_GATE)
#define SZ_AB    (BC_*H_*WO_)              // 4669440
#define OFF_B    (OFF_A + SZ_AB)
#define OFF_SB   (OFF_B + SZ_AB)
#define SZ_SB    (4*BC_*WO_*WO_)           // 18483200
#define OFF_Y    (OFF_SB + SZ_SB)
#define SZ_Y     (B_*CIN_*HW_)             // 18874368

// ---------------- filter tables ----------------
__constant__ float c_fir[4][30] = {
  // db4 (L=8)
  {-0.0105974017850021f, 0.0328830116668852f, 0.0308413818355607f, -0.1870348117188811f,
   -0.0279837694169839f, 0.6308807679295904f, 0.7148465705529155f, 0.2303778133088964f,
   0,0,0,0,0,0,0,0,0,0,0,0,0,0,0,0,0,0,0,0,0,0},
  // db6 (L=12)
  {0.00107730108499558f, -0.00477725751101065f, -0.0005538422009938f, 0.03158203931748603f,
   0.02752286553030533f, -0.0975016055873225f, -0.12976686756709563f, 0.22626469396544f,
   0.3152503517092432f, -0.7511339080210959f, 0.4946238903984534f, 0.1115407433501095f,
   0,0,0,0,0,0,0,0,0,0,0,0,0,0,0,0,0,0},
  // sym6 (L=12)
  {-0.007800708325034148f, 0.001767711864242804f, 0.04472490177066578f, -0.02106029251230056f,
   -0.0726375227866f, 0.3379294217282401f, 0.787641141030194f, 0.4910559419267466f,
   -0.048311742585632f, -0.1179901111484105f, 0.00349071208421747f, 0.01540410932702737f,
   0,0,0,0,0,0,0,0,0,0,0,0,0,0,0,0,0,0},
  // coif5 (L=30)
  {-3.459977283621256e-05f, -7.098330313814114e-05f, 0.0004662169601128863f, 0.001117518770890601f,
   -0.002574517688750223f, -0.00900797613666158f, 0.015880544863615904f, 0.03455502757306163f,
   -0.08230192710688598f, -0.07179982161931202f, 0.42848347637761874f, 0.7937772226256206f,
   0.4051769024096169f, -0.06112339000267287f, -0.06577191128185562f, 0.023452696141836267f,
   0.007782596427325418f, -0.003793512864491014f, -0.0002606761356811993f, 0.000107502882505652f,
   1.10319778524429e-05f, -5.520763127949e-06f, -1.0682196848076e-06f, 5.236425333584e-07f,
   1.125098976034e-07f, -5.417490769329e-08f, -8.8631e-09f, 4.2921e-09f, 6.7e-10f, -3.2e-10f}
};
__constant__ int c_firL[4] = {8, 12, 12, 30};
__constant__ float c_l53[2] = {-0.5f, 0.25f};
__constant__ float c_l97[4] = {-1.586134342f, -0.05298011854f, 0.8829110762f, 0.4435068522f};

__device__ __forceinline__ int reflect_i(int i, int n) {
  if (i < 0) i = -i;
  if (i >= n) i = 2*n - 2 - i;
  return i;
}

// ---------------- gate: per-pixel MLP + softmax + renorm ----------------
__global__ void gate_kernel(const float* __restrict__ x,
                            const float* __restrict__ w1, const float* __restrict__ b1,
                            const float* __restrict__ w2, const float* __restrict__ b2,
                            float* __restrict__ gate) {
  int idx = blockIdx.x * blockDim.x + threadIdx.x;
  if (idx >= B_ * HW_) return;
  int p = idx % HW_;
  int b = idx / HW_;
  float xin[C_];
  #pragma unroll
  for (int c = 0; c < C_; ++c) xin[c] = x[((b*C_ + c) * HW_) + p];
  float h[8];
  #pragma unroll
  for (int j = 0; j < 8; ++j) {
    float s = b1[j];
    #pragma unroll
    for (int c = 0; c < C_; ++c) s += w1[j*C_ + c] * xin[c];
    h[j] = fmaxf(s, 0.0f);
  }
  float lg[K_];
  float m = -3.4e38f;
  #pragma unroll
  for (int k = 0; k < K_; ++k) {
    float s = b2[k];
    #pragma unroll
    for (int j = 0; j < 8; ++j) s += w2[k*8 + j] * h[j];
    lg[k] = s;
    m = fmaxf(m, s);
  }
  float den = 0.0f;
  #pragma unroll
  for (int k = 0; k < K_; ++k) { lg[k] = __expf(lg[k] - m); den += lg[k]; }
  float inv = 1.0f / den;
  float s2 = 0.0f;
  #pragma unroll
  for (int k = 0; k < K_; ++k) { lg[k] *= inv; s2 += lg[k]; }
  float inv2 = 1.0f / (s2 + 1e-12f);
  #pragma unroll
  for (int k = 0; k < K_; ++k) gate[((b*K_ + k) * HW_) + p] = lg[k] * inv2;
}

// ---------------- zero fill ----------------
__global__ void zero_kernel(float* __restrict__ p, long n) {
  long stride = (long)gridDim.x * blockDim.x;
  for (long i = blockIdx.x * (long)blockDim.x + threadIdx.x; i < n; i += stride) p[i] = 0.0f;
}

// ---------------- FIR row pass (W direction), 192 -> 190 ----------------
__global__ void fir_wpass(const float* __restrict__ x, float* __restrict__ Lw,
                          float* __restrict__ Hw, int cand) {
  int idx = blockIdx.x * blockDim.x + threadIdx.x;
  if (idx >= BC_ * H_ * WO_) return;
  int j = idx % WO_;
  int t = idx / WO_;
  int h = t % H_;
  int img = t / H_;
  const int L = c_firL[cand];
  const int pa = (L - 1) / 2 - 1;   // even L
  float ssq = 0.0f;
  for (int u = 0; u < L; ++u) { float v = c_fir[cand][u]; ssq += v * v; }
  float inv = 1.0f / (sqrtf(ssq) + 1e-12f);
  const float* row = x + (long)(img*H_ + h) * W_;
  float al = 0.0f, ah = 0.0f;
  for (int u = 0; u < L; ++u) {
    float lo = c_fir[cand][u] * inv;
    float hi = ((u & 1) ? -1.0f : 1.0f) * c_fir[cand][L-1-u] * inv;
    float v = row[reflect_i(j + u - pa, W_)];
    al += lo * v;
    ah += hi * v;
  }
  Lw[(long)(img*H_ + h) * WO_ + j] = al;
  Hw[(long)(img*H_ + h) * WO_ + j] = ah;
}

// ---------------- FIR column pass (H direction), -> 190x190 subbands ----------------
__global__ void fir_hpass(const float* __restrict__ Lw, const float* __restrict__ Hw,
                          float* __restrict__ sb, int cand) {
  int idx = blockIdx.x * blockDim.x + threadIdx.x;
  if (idx >= BC_ * WO_ * WO_) return;
  int q = idx % WO_;
  int t = idx / WO_;
  int r = t % WO_;
  int img = t / WO_;
  const int L = c_firL[cand];
  const int pa = (L - 1) / 2 - 1;
  float ssq = 0.0f;
  for (int u = 0; u < L; ++u) { float v = c_fir[cand][u]; ssq += v * v; }
  float inv = 1.0f / (sqrtf(ssq) + 1e-12f);
  float LL = 0.f, LH = 0.f, HL = 0.f, HH = 0.f;
  for (int u = 0; u < L; ++u) {
    float lo = c_fir[cand][u] * inv;
    float hi = ((u & 1) ? -1.0f : 1.0f) * c_fir[cand][L-1-u] * inv;
    int rr = reflect_i(r + u - pa, H_);
    float vl = Lw[(long)(img*H_ + rr) * WO_ + q];
    float vh = Hw[(long)(img*H_ + rr) * WO_ + q];
    LL += lo * vl; LH += hi * vl; HL += lo * vh; HH += hi * vh;
  }
  long ps = (long)WO_ * WO_;
  long base = (long)img * ps + (long)r * WO_ + q;
  sb[0*BC_*ps + base] = LL;
  sb[1*BC_*ps + base] = LH;
  sb[2*BC_*ps + base] = HL;
  sb[3*BC_*ps + base] = HH;
}

// ---------------- lifting row pass: x(192 wide) -> A,D (96 wide) ----------------
__global__ void lift_wpass(const float* __restrict__ x, float* __restrict__ A,
                           float* __restrict__ D, const float* __restrict__ sAp,
                           const float* __restrict__ sDp, int lcand) {
  __shared__ float e[HN_], o[HN_];
  int blk = blockIdx.x;           // img*H_ + row
  int img = blk / H_;
  int row = blk % H_;
  int j = threadIdx.x;            // 0..95
  const float* rowp = x + (long)(img*H_ + row) * W_;
  e[j] = rowp[2*j];
  o[j] = rowp[2*j + 1];
  __syncthreads();
  int ns = lcand ? 4 : 2;
  for (int s = 0; s < ns; ++s) {
    float cf = lcand ? c_l97[s] : c_l53[s];
    if ((s & 1) == 0) o[j] += cf * 0.5f * (e[reflect_i(j-1, HN_)] + e[reflect_i(j+1, HN_)]);
    else              e[j] += cf * 0.5f * (o[reflect_i(j-1, HN_)] + o[reflect_i(j+1, HN_)]);
    __syncthreads();
  }
  A[(long)(img*H_ + row) * HN_ + j] = (*sAp) * e[j];
  D[(long)(img*H_ + row) * HN_ + j] = (*sDp) * o[j];
}

// ---------------- lifting column pass: A/D (192x96) -> 96x96 subbands ----------------
__global__ void lift_hpass(const float* __restrict__ A, const float* __restrict__ D,
                           float* __restrict__ sb, const float* __restrict__ sAp,
                           const float* __restrict__ sDp, int lcand) {
  __shared__ float e[HN_], o[HN_];
  int blk = blockIdx.x;           // (src*BC_ + img)*HN_ + col
  int col = blk % HN_;
  int img = (blk / HN_) % BC_;
  int src = blk / (HN_ * BC_);
  const float* In = src ? D : A;
  int i = threadIdx.x;            // 0..95
  e[i] = In[(long)(img*H_ + 2*i) * HN_ + col];
  o[i] = In[(long)(img*H_ + 2*i + 1) * HN_ + col];
  __syncthreads();
  int ns = lcand ? 4 : 2;
  for (int s = 0; s < ns; ++s) {
    float cf = lcand ? c_l97[s] : c_l53[s];
    if ((s & 1) == 0) o[i] += cf * 0.5f * (e[reflect_i(i-1, HN_)] + e[reflect_i(i+1, HN_)]);
    else              e[i] += cf * 0.5f * (o[reflect_i(i-1, HN_)] + o[reflect_i(i+1, HN_)]);
    __syncthreads();
  }
  long ps = (long)HN_ * HN_;
  int s0 = src * 2;               // A -> LL,LH ; D -> HL,HH
  sb[((long)(s0  ) * BC_ + img) * ps + (long)i * HN_ + col] = (*sAp) * e[i];
  sb[((long)(s0+1) * BC_ + img) * ps + (long)i * HN_ + col] = (*sDp) * o[i];
}

// ---------------- bilinear upsample (half-pixel, edge clamp) + gated accumulate into Y ----
__global__ void up_blend(const float* __restrict__ sb, int sn,
                         const float* __restrict__ gate, const float* __restrict__ hfp,
                         float* __restrict__ Y, int kcand) {
  int idx = blockIdx.x * blockDim.x + threadIdx.x;
  if (idx >= BC_ * HW_) return;
  int p = idx % HW_;
  int img = idx / HW_;
  int i = p / W_, j = p % W_;
  int b = img / C_, c = img % C_;
  float scale = (float)sn / (float)H_;
  float si = (i + 0.5f) * scale - 0.5f;
  float sj = (j + 0.5f) * scale - 0.5f;
  int i0 = (int)floorf(si), j0 = (int)floorf(sj);
  float fi = si - (float)i0, fj = sj - (float)j0;
  int i0c = max(i0, 0),      j0c = max(j0, 0);
  int i1c = min(i0 + 1, sn - 1), j1c = min(j0 + 1, sn - 1);
  float w = gate[((long)(b*K_ + kcand)) * HW_ + p];
  float hf = *hfp;
  long ps = (long)sn * sn;
  #pragma unroll
  for (int s = 0; s < 4; ++s) {
    const float* pl = sb + ((long)s * BC_ + img) * ps;
    float v00 = pl[(long)i0c * sn + j0c], v01 = pl[(long)i0c * sn + j1c];
    float v10 = pl[(long)i1c * sn + j0c], v11 = pl[(long)i1c * sn + j1c];
    float v = (1.0f - fi) * ((1.0f - fj) * v00 + fj * v01)
            +         fi  * ((1.0f - fj) * v10 + fj * v11);
    if (s) v *= hf;
    Y[((long)(b * CIN_ + c * 4 + s)) * HW_ + p] += w * v;
  }
}

// ---------------- projection: out = proj_w (32x128) @ Y (128 x pixels) + b ----------------
// WMMA bf16 16x16x32, fp32 accumulate. One wave holds the full A fragment set
// (proj_w rows for its 16 output channels, 4 K-chunks) in VGPRs and sweeps 4
// consecutive 16-pixel N-tiles against it: 16 v_wmma per wave, A loaded once.
// Native (__bf16) converts keep the VALU pipe free of manual rounding code.
typedef __attribute__((ext_vector_type(16))) __bf16 v16bf;
typedef __attribute__((ext_vector_type(8)))  float  v8f;

__global__ void proj_wmma(const float* __restrict__ Y, const float* __restrict__ pw,
                          const float* __restrict__ pb, float* __restrict__ out) {
  const int NG = HW_ / 64;                       // 576 groups of 4 pixel tiles
  int wave = (blockIdx.x * blockDim.x + threadIdx.x) >> 5;
  int lane = threadIdx.x & 31;
  int g  = wave % NG;
  int mt = (wave / NG) & 1;                      // 2 tiles of 16 output channels
  int b  = wave / (2 * NG);
  int pbase = g * 64;
  int lm = lane & 15;
  int half = lane >> 4;
  int gA = half * 8;                             // A lane K split {0-7,16-23}/{8-15,24-31}
  int gB = half * 8;                             // B lane K split {0-7,16-23}/{8-15,24-31}

  // A fragments: load proj_w once per wave (shared by all 4 N-tiles).
  v16bf a[4];
  #pragma unroll
  for (int ks = 0; ks < 4; ++ks) {
    int k0 = ks * 32;
    #pragma unroll
    for (int t = 0; t < 8; ++t) {
      a[ks][t]     = (__bf16)pw[(mt*16 + lm) * CIN_ + k0 + gA + t];
      a[ks][t + 8] = (__bf16)pw[(mt*16 + lm) * CIN_ + k0 + 16 + gA + t];
    }
  }

  #pragma unroll
  for (int nt = 0; nt < 4; ++nt) {
    int p0 = pbase + nt * 16;
    v8f acc = {};
    #pragma unroll
    for (int ks = 0; ks < 4; ++ks) {
      int k0 = ks * 32;
      v16bf bm;
      #pragma unroll
      for (int e = 0; e < 16; ++e)
        bm[e] = (__bf16)Y[((long)(b * CIN_ + k0 + gB + e + (e >= 8 ? 8 : 0))) * HW_ + p0 + lm];
      acc = __builtin_amdgcn_wmma_f32_16x16x32_bf16(false, a[ks], false, bm,
                                                    (short)0, acc, false, false);
    }
    #pragma unroll
    for (int r = 0; r < 8; ++r) {
      int oc = mt * 16 + half * 8 + r;           // C/D layout: lanes 16-31 hold M=8..15
      out[((long)(b * COUT_ + oc)) * HW_ + p0 + lm] = acc[r] + pb[oc];
    }
  }
}

// ---------------- orchestration ----------------
extern "C" void kernel_launch(void* const* d_in, const int* in_sizes, int n_in,
                              void* d_out, int out_size, void* d_ws, size_t ws_size,
                              hipStream_t stream) {
  (void)in_sizes; (void)n_in; (void)out_size; (void)ws_size;
  const float* x    = (const float*)d_in[0];
  const float* gw1  = (const float*)d_in[1];
  const float* gb1  = (const float*)d_in[2];
  const float* gw2  = (const float*)d_in[3];
  const float* gb2  = (const float*)d_in[4];
  const float* s53A = (const float*)d_in[5];
  const float* s53D = (const float*)d_in[6];
  const float* s97A = (const float*)d_in[7];
  const float* s97D = (const float*)d_in[8];
  const float* hf   = (const float*)d_in[9];
  const float* pw   = (const float*)d_in[10];
  const float* pb   = (const float*)d_in[11];
  float* out = (float*)d_out;

  float* ws   = (float*)d_ws;     // requires ~190.3 MB of scratch
  float* gate = ws + OFF_GATE;
  float* bufA = ws + OFF_A;
  float* bufB = ws + OFF_B;
  float* sb   = ws + OFF_SB;
  float* Y    = ws + OFF_Y;

  gate_kernel<<<(B_*HW_ + 255)/256, 256, 0, stream>>>(x, gw1, gb1, gw2, gb2, gate);
  zero_kernel<<<4096, 256, 0, stream>>>(Y, (long)SZ_Y);

  // FIR candidates: k = 0..3 (db4, db6, sym6, coif5)
  for (int cand = 0; cand < 4; ++cand) {
    fir_wpass<<<(BC_*H_*WO_ + 255)/256, 256, 0, stream>>>(x, bufA, bufB, cand);
    fir_hpass<<<(BC_*WO_*WO_ + 255)/256, 256, 0, stream>>>(bufA, bufB, sb, cand);
    up_blend <<<(BC_*HW_ + 255)/256, 256, 0, stream>>>(sb, WO_, gate, hf, Y, cand);
  }
  // bior5/3: k = 4
  lift_wpass<<<BC_*H_,      HN_, 0, stream>>>(x, bufA, bufB, s53A, s53D, 0);
  lift_hpass<<<2*BC_*HN_,   HN_, 0, stream>>>(bufA, bufB, sb, s53A, s53D, 0);
  up_blend  <<<(BC_*HW_ + 255)/256, 256, 0, stream>>>(sb, HN_, gate, hf, Y, 4);
  // bior9/7: k = 5
  lift_wpass<<<BC_*H_,      HN_, 0, stream>>>(x, bufA, bufB, s97A, s97D, 1);
  lift_hpass<<<2*BC_*HN_,   HN_, 0, stream>>>(bufA, bufB, sb, s97A, s97D, 1);
  up_blend  <<<(BC_*HW_ + 255)/256, 256, 0, stream>>>(sb, HN_, gate, hf, Y, 5);

  // Projection GEMM on the matrix pipe: 4*2*576 = 4608 waves, 8 waves/block.
  proj_wmma<<<(B_ * 2 * (HW_/64)) / 8, 256, 0, stream>>>(Y, pw, pb, out);
}
